// AtencaoLatenteMultiCabeca_22333829939900
// MI455X (gfx1250) — matrix operand order repaired and checked by reading, not compiled, not run on hardware
//
#include <hip/hip_runtime.h>

// ---------------- CDNA5 WMMA types ----------------
typedef __attribute__((ext_vector_type(16))) __bf16 v16bf;
typedef __attribute__((ext_vector_type(8)))  float  v8f;

union FragB { unsigned short u[16]; uint4 q[2]; v16bf v; };

__device__ __forceinline__ unsigned short f2bf(float f) {
  unsigned int u = __float_as_uint(f);
  u += 0x7FFFu + ((u >> 16) & 1u);          // round-to-nearest-even
  return (unsigned short)(u >> 16);
}
__device__ __forceinline__ float bf2f(unsigned short h) {
  return __uint_as_float(((unsigned int)h) << 16);
}

// ---------------- problem constants ----------------
constexpr int CB = 2;
constexpr int CS = 2048;
constexpr int CDIM = 2048;
constexpr int CH = 16;
constexpr int CKVR = 512;        // KV_RANK
constexpr int CNOPE = 128;
constexpr int CROPE = 64;
constexpr int CVDIM = 128;
constexpr int CQK = CNOPE + CROPE;   // 192
constexpr int CM = CB * CS;          // 4096 tokens
constexpr float CSCALE = 0.07216878364870323f;  // 192^-0.5

// ---------------- f32 -> bf16 convert ----------------
__global__ void f32_to_bf16(const float* __restrict__ in,
                            unsigned short* __restrict__ out, long long n) {
  long long i = (long long)blockIdx.x * blockDim.x + threadIdx.x;
  if (i < n) out[i] = f2bf(in[i]);
}

// ---------------- wb_k transpose: [H][128 d][512 c] f32 -> [H][512 c][128 d] bf16
__global__ void transpose_wbk(const float* __restrict__ Wkvb,      // [H*256][512]
                              unsigned short* __restrict__ WbkT) { // [H][512][128]
  long long idx = (long long)blockIdx.x * blockDim.x + threadIdx.x;
  if (idx >= (long long)CH * CKVR * CNOPE) return;
  int d = (int)(idx % CNOPE);
  int c = (int)((idx / CNOPE) % CKVR);
  int h = (int)(idx / ((long long)CNOPE * CKVR));
  WbkT[idx] = f2bf(Wkvb[((long long)h * (CNOPE + CVDIM) + d) * CKVR + c]);
}

// ---------------- generic bf16 WMMA GEMM (NT only) ----------------
// C[M,N] = A[M,K] * B^T;  A row-major lda=K, B row-major [N x K] ldb=K.
// c_f32: 1 -> float C, 0 -> bf16 C.  Batched via gridDim.z with element strides.
// LDS B tile stored N-major so every fragment load is ds_load_b128.
__global__ __launch_bounds__(256) void gemm_bf16(
    const unsigned short* __restrict__ A,
    const unsigned short* __restrict__ Bm,
    void* __restrict__ Cm,
    int N, int K, int lda, int ldb, int ldc,
    int c_f32,
    long long sA, long long sB, long long sC)
{
  __shared__ unsigned short As[128 * 32];   // [m][k]
  __shared__ unsigned short Bs[64 * 32];    // [n][k]

  const int tid  = threadIdx.x;
  const int lane = tid & 31;
  const int wave = tid >> 5;          // 0..7
  const int wm   = wave >> 1;         // 0..3 -> 32-row slab
  const int wn   = wave & 1;          // 0..1 -> 32-col slab
  const int l15  = lane & 15;
  const int hl   = (lane < 16) ? 0 : 1;

  const int m0 = blockIdx.x * 128;
  const int n0 = blockIdx.y * 64;
  const long long batch = blockIdx.z;
  A  += batch * sA;
  Bm += batch * sB;

  v8f acc[2][2] = {};

  for (int k0 = 0; k0 < K; k0 += 32) {
    // ---- stage A tile 128x32 (each thread: 16 contiguous bf16)
    {
      int r  = tid >> 1;
      int kk = (tid & 1) * 16;
      const unsigned short* src = A + (long long)(m0 + r) * lda + (k0 + kk);
      const uint4* s4 = (const uint4*)src;
      *(uint4*)&As[r * 32 + kk]     = s4[0];
      *(uint4*)&As[r * 32 + kk + 8] = s4[1];
    }
    // ---- stage B tile into Bs[n][k] (contiguous K from NT layout)
    {
      int n  = tid >> 2;          // 0..63
      int kk = (tid & 3) * 8;     // 0,8,16,24
      const unsigned short* src = Bm + (long long)(n0 + n) * ldb + (k0 + kk);
      *(uint4*)&Bs[n * 32 + kk] = *(const uint4*)src;
    }
    __syncthreads();

    // ---- fragments in contiguous k order, K = hl*16 + s, same for A and B (all b128 loads)
    FragB a[2], b[2];
    #pragma unroll
    for (int ms = 0; ms < 2; ++ms) {
      int row = wm * 32 + ms * 16 + l15;
      a[ms].q[0] = *(const uint4*)&As[row * 32 + hl * 16];       // K = hl*16 + 0..7
      a[ms].q[1] = *(const uint4*)&As[row * 32 + hl * 16 + 8];   // K = hl*16 + 8..15
    }
    #pragma unroll
    for (int ns = 0; ns < 2; ++ns) {
      int col = wn * 32 + ns * 16 + l15;
      const uint4* bp = (const uint4*)&Bs[col * 32 + hl * 16];   // K = hl*16 + 0..15
      b[ns].q[0] = bp[0];
      b[ns].q[1] = bp[1];
    }
    #pragma unroll
    for (int ms = 0; ms < 2; ++ms)
      #pragma unroll
      for (int ns = 0; ns < 2; ++ns)
        acc[ms][ns] = __builtin_amdgcn_wmma_f32_16x16x32_bf16(
            false, a[ms].v, false, b[ns].v, (short)0, acc[ms][ns], false, false);
    __syncthreads();
  }

  // ---- store C (VGPR r: lanes0-15 row=r, lanes16-31 row=8+r; lane&15 = col)
  #pragma unroll
  for (int ms = 0; ms < 2; ++ms) {
    #pragma unroll
    for (int ns = 0; ns < 2; ++ns) {
      int colg = n0 + wn * 32 + ns * 16 + l15;
      #pragma unroll
      for (int r = 0; r < 8; ++r) {
        int rowg = m0 + wm * 32 + ms * 16 + r + hl * 8;
        long long off = batch * sC + (long long)rowg * ldc + colg;
        float val = acc[ms][ns][r];
        if (c_f32) ((float*)Cm)[off] = val;
        else       ((unsigned short*)Cm)[off] = f2bf(val);
      }
    }
  }
}

// ---------------- q split + RoPE (+ SCALE folded in) ----------------
__global__ void prep_q(const unsigned short* __restrict__ Qb,   // [M][H*QK] bf16
                       const float* __restrict__ cosT, const float* __restrict__ sinT,
                       unsigned short* __restrict__ Qnope,      // [H][M][128]
                       unsigned short* __restrict__ Qpe)        // [H][M][64]
{
  long long idx = (long long)blockIdx.x * blockDim.x + threadIdx.x;
  if (idx >= (long long)CM * CH) return;
  int h = (int)(idx % CH);
  long long m = idx / CH;
  int s = (int)(m % CS);
  const unsigned short* src = Qb + m * (CH * CQK) + h * CQK;
  unsigned short* dn = Qnope + ((long long)h * CM + m) * CNOPE;
  #pragma unroll 4
  for (int d = 0; d < CNOPE; ++d) dn[d] = f2bf(CSCALE * bf2f(src[d]));
  unsigned short* dp = Qpe + ((long long)h * CM + m) * CROPE;
  const float* cr = cosT + (long long)s * (CROPE / 2);
  const float* sr = sinT + (long long)s * (CROPE / 2);
  #pragma unroll 4
  for (int i = 0; i < CROPE / 2; ++i) {
    float xe = bf2f(src[CNOPE + 2 * i]);
    float xo = bf2f(src[CNOPE + 2 * i + 1]);
    float c = cr[i], sn = sr[i];
    dp[2 * i]     = f2bf(CSCALE * (xe * c - xo * sn));
    dp[2 * i + 1] = f2bf(CSCALE * (xe * sn + xo * c));
  }
}

// ---------------- kv split: rmsnorm(c_kv), rope(k_pe) ----------------
__global__ __launch_bounds__(256) void prep_kv(
    const unsigned short* __restrict__ KVb,   // [M][576] bf16
    const float* __restrict__ w,              // [512]
    const float* __restrict__ cosT, const float* __restrict__ sinT,
    unsigned short* __restrict__ CKV,         // [M][512]
    unsigned short* __restrict__ KPE)         // [M][64]
{
  __shared__ float red[256];
  long long m = blockIdx.x;
  int s = (int)(m % CS);
  const unsigned short* src = KVb + m * (CKVR + CROPE);
  float local = 0.f;
  for (int c = threadIdx.x; c < CKVR; c += 256) {
    float v = bf2f(src[c]);
    local += v * v;
  }
  red[threadIdx.x] = local;
  __syncthreads();
  for (int o = 128; o > 0; o >>= 1) {
    if (threadIdx.x < o) red[threadIdx.x] += red[threadIdx.x + o];
    __syncthreads();
  }
  float rms = rsqrtf(red[0] / (float)CKVR + 1e-6f);
  for (int c = threadIdx.x; c < CKVR; c += 256)
    CKV[m * CKVR + c] = f2bf(bf2f(src[c]) * rms * w[c]);
  if (threadIdx.x < CROPE / 2) {
    int i = threadIdx.x;
    float xe = bf2f(src[CKVR + 2 * i]);
    float xo = bf2f(src[CKVR + 2 * i + 1]);
    float c = cosT[(long long)s * (CROPE / 2) + i];
    float sn = sinT[(long long)s * (CROPE / 2) + i];
    KPE[m * CROPE + 2 * i]     = f2bf(xe * c - xo * sn);
    KPE[m * CROPE + 2 * i + 1] = f2bf(xe * sn + xo * c);
  }
}

// ---------------- fused MLA attention (flash-style, bf16 WMMA) ----------------
// block = 4 waves; one (batch, head, 16-query tile) per block.
// Score K-dim (576) is round-robined across waves, reduced through LDS.
// wave w owns o_lat columns [w*128, w*128+128).
__global__ __launch_bounds__(128) void mla_attention(
    const unsigned short* __restrict__ Qlat,   // [H][M][512], pre-scaled
    const unsigned short* __restrict__ Qpe,    // [H][M][64],  pre-scaled
    const unsigned short* __restrict__ CKV,    // [M][512]
    const unsigned short* __restrict__ KPE,    // [M][64]
    unsigned short* __restrict__ Olat)         // [H][M][512]
{
  constexpr int DKQ = CKVR + CROPE;            // 576
  __shared__ unsigned short Qs[16 * DKQ];      // 18 KB
  __shared__ unsigned short Ks[32 * DKQ];      // 36 KB
  __shared__ unsigned short Ps[16 * 32];       //  1 KB
  __shared__ float Sred[4][16][32];            //  8 KB partial scores

  const int tid  = threadIdx.x;
  const int lane = tid & 31;
  const int wave = tid >> 5;       // 0..3
  const int l15  = lane & 15;
  const int hl   = (lane < 16) ? 0 : 1;

  const int s0 = blockIdx.x * 16;
  const int h  = blockIdx.y;
  const int b  = blockIdx.z;
  const long long mbase = (long long)b * CS + s0;

  // stage 16x576 Q tile (q_lat || q_pe)
  for (int base = tid * 8; base < 16 * DKQ; base += 128 * 8) {
    int r = base / DKQ, c = base % DKQ;
    const unsigned short* src =
        (c < CKVR) ? &Qlat[((long long)h * CM + mbase + r) * CKVR + c]
                   : &Qpe[((long long)h * CM + mbase + r) * CROPE + (c - CKVR)];
    *(uint4*)&Qs[base] = *(const uint4*)src;
  }
  __syncthreads();

  float m_run[8], l_run[8];
  #pragma unroll
  for (int r = 0; r < 8; ++r) { m_run[r] = -1e30f; l_run[r] = 0.f; }
  v8f acc[8] = {};

  const int ntile = (s0 + 16 + 31) / 32;
  for (int t = 0; t < ntile; ++t) {
    const int t0 = t * 32;
    // stage 32x576 key tile (c_kv || k_pe)
    for (int base = tid * 8; base < 32 * DKQ; base += 128 * 8) {
      int r = base / DKQ, c = base % DKQ;
      long long tg = (long long)b * CS + t0 + r;
      const unsigned short* src =
          (c < CKVR) ? &CKV[tg * CKVR + c] : &KPE[tg * CROPE + (c - CKVR)];
      *(uint4*)&Ks[base] = *(const uint4*)src;
    }
    // prefetch next key tile (lowers to global_prefetch_b8)
    if (t + 1 < ntile) {
      long long nrow = (long long)b * CS + t0 + 32 + lane;
      __builtin_prefetch(&CKV[nrow * CKVR + wave * 128], 0, 1);
    }
    __syncthreads();

    // partial scores: this wave's K-chunks only (round-robin over 18 chunks)
    v8f sA0 = {}, sA1 = {};
    for (int kc = wave; kc < DKQ / 32; kc += 4) {
      int kb = kc * 32;
      FragB a, b0, b1;
      a.q[0] = *(const uint4*)&Qs[l15 * DKQ + kb + hl * 16];
      a.q[1] = *(const uint4*)&Qs[l15 * DKQ + kb + hl * 16 + 8];
      const uint4* bq0 = (const uint4*)&Ks[l15 * DKQ + kb + hl * 16];
      b0.q[0] = bq0[0]; b0.q[1] = bq0[1];
      const uint4* bq1 = (const uint4*)&Ks[(16 + l15) * DKQ + kb + hl * 16];
      b1.q[0] = bq1[0]; b1.q[1] = bq1[1];
      sA0 = __builtin_amdgcn_wmma_f32_16x16x32_bf16(false, a.v, false, b0.v, (short)0, sA0, false, false);
      sA1 = __builtin_amdgcn_wmma_f32_16x16x32_bf16(false, a.v, false, b1.v, (short)0, sA1, false, false);
    }
    #pragma unroll
    for (int r = 0; r < 8; ++r) {
      Sred[wave][r + hl * 8][l15]      = sA0[r];
      Sred[wave][r + hl * 8][16 + l15] = sA1[r];
    }
    __syncthreads();

    // sum partials, causal mask + online softmax (rows r+8*hl live in lane-halves)
    float p0[8], p1[8], mt[8];
    #pragma unroll
    for (int r = 0; r < 8; ++r) {
      int rr = r + hl * 8;
      float v0 = Sred[0][rr][l15] + Sred[1][rr][l15] +
                 Sred[2][rr][l15] + Sred[3][rr][l15];
      float v1 = Sred[0][rr][16 + l15] + Sred[1][rr][16 + l15] +
                 Sred[2][rr][16 + l15] + Sred[3][rr][16 + l15];
      int rowg = s0 + rr;
      if (t0 + l15 > rowg)      v0 = -1e30f;
      if (t0 + 16 + l15 > rowg) v1 = -1e30f;
      p0[r] = v0; p1[r] = v1;
      mt[r] = fmaxf(v0, v1);
    }
    #pragma unroll
    for (int off = 8; off >= 1; off >>= 1)
      #pragma unroll
      for (int r = 0; r < 8; ++r)
        mt[r] = fmaxf(mt[r], __shfl_xor(mt[r], off, 32));

    float sc[8], st[8];
    #pragma unroll
    for (int r = 0; r < 8; ++r) {
      float mn = fmaxf(m_run[r], mt[r]);
      sc[r] = __expf(m_run[r] - mn);
      m_run[r] = mn;
      p0[r] = __expf(p0[r] - mn);
      p1[r] = __expf(p1[r] - mn);
      st[r] = p0[r] + p1[r];
    }
    #pragma unroll
    for (int off = 8; off >= 1; off >>= 1)
      #pragma unroll
      for (int r = 0; r < 8; ++r)
        st[r] += __shfl_xor(st[r], off, 32);
    #pragma unroll
    for (int r = 0; r < 8; ++r) l_run[r] = l_run[r] * sc[r] + st[r];
    #pragma unroll
    for (int j = 0; j < 8; ++j)
      #pragma unroll
      for (int r = 0; r < 8; ++r)
        acc[j][r] *= sc[r];

    // P re-layout C->A through LDS (identical values in every wave; wave0 writes)
    if (wave == 0) {
      #pragma unroll
      for (int r = 0; r < 8; ++r) {
        Ps[(r + hl * 8) * 32 + l15]      = f2bf(p0[r]);
        Ps[(r + hl * 8) * 32 + 16 + l15] = f2bf(p1[r]);
      }
    }
    __syncthreads();

    FragB ap;
    ap.q[0] = *(const uint4*)&Ps[l15 * 32 + hl * 16];
    ap.q[1] = *(const uint4*)&Ps[l15 * 32 + hl * 16 + 8];
    const int cw = wave * 128;
    #pragma unroll
    for (int j = 0; j < 8; ++j) {
      FragB bv;   // V^T gather (stride DKQ): inherent transpose, scalar u16 loads
      #pragma unroll
      for (int qq = 0; qq < 16; ++qq) {
        int kk = qq + hl * 16;
        bv.u[qq] = Ks[kk * DKQ + cw + j * 16 + l15];
      }
      acc[j] = __builtin_amdgcn_wmma_f32_16x16x32_bf16(
          false, ap.v, false, bv.v, (short)0, acc[j], false, false);
    }
    __syncthreads();   // protect Ks/Ps/Sred before next tile
  }

  // normalize and write o_lat (bf16)
  #pragma unroll
  for (int r = 0; r < 8; ++r) {
    float inv = 1.0f / l_run[r];
    int rowg = r + hl * 8;
    long long m = mbase + rowg;
    #pragma unroll
    for (int j = 0; j < 8; ++j) {
      int c = wave * 128 + j * 16 + l15;
      Olat[((long long)h * CM + m) * CKVR + c] = f2bf(acc[j][r] * inv);
    }
  }
}

// ---------------- host: orchestration ----------------
extern "C" void kernel_launch(void* const* d_in, const int* in_sizes, int n_in,
                              void* d_out, int out_size, void* d_ws, size_t ws_size,
                              hipStream_t stream) {
  (void)in_sizes; (void)n_in; (void)out_size; (void)ws_size;
  const float* x    = (const float*)d_in[0];
  const float* wq   = (const float*)d_in[1];
  const float* wkva = (const float*)d_in[2];
  const float* kvw  = (const float*)d_in[3];
  const float* wkvb = (const float*)d_in[4];
  const float* wo   = (const float*)d_in[5];
  const float* cosT = (const float*)d_in[6];
  const float* sinT = (const float*)d_in[7];

  char* p = (char*)d_ws;
  auto take = [&](long long elems) {
    char* r = p;
    p += ((elems * 2 + 255) / 256) * 256;     // bf16 elements, 256B aligned
    return (unsigned short*)r;
  };
  unsigned short* xb    = take((long long)CM * CDIM);
  unsigned short* wqb   = take((long long)CH * CQK * CDIM);
  unsigned short* wkvab = take((long long)(CKVR + CROPE) * CDIM);
  unsigned short* wkvbb = take((long long)CH * (CNOPE + CVDIM) * CKVR);
  unsigned short* wbkT  = take((long long)CH * CKVR * CNOPE);   // transposed wb_k
  unsigned short* wob   = take((long long)CDIM * CH * CVDIM);
  unsigned short* qb    = take((long long)CM * CH * CQK);
  unsigned short* kvb   = take((long long)CM * (CKVR + CROPE));
  unsigned short* qnope = take((long long)CH * CM * CNOPE);
  unsigned short* qpe   = take((long long)CH * CM * CROPE);
  unsigned short* ckv   = take((long long)CM * CKVR);
  unsigned short* kpe   = take((long long)CM * CROPE);
  unsigned short* qlat  = take((long long)CH * CM * CKVR);
  unsigned short* olat  = take((long long)CH * CM * CKVR);
  unsigned short* ob    = take((long long)CM * CH * CVDIM);

  auto cvt = [&](const float* src, unsigned short* dst, long long n) {
    f32_to_bf16<<<dim3((unsigned)((n + 255) / 256)), dim3(256), 0, stream>>>(src, dst, n);
  };
  cvt(x,    xb,    (long long)CM * CDIM);
  cvt(wq,   wqb,   (long long)CH * CQK * CDIM);
  cvt(wkva, wkvab, (long long)(CKVR + CROPE) * CDIM);
  cvt(wkvb, wkvbb, (long long)CH * (CNOPE + CVDIM) * CKVR);
  cvt(wo,   wob,   (long long)CDIM * CH * CVDIM);
  transpose_wbk<<<dim3((CH * CKVR * CNOPE) / 256), dim3(256), 0, stream>>>(wkvb, wbkT);

  // q = x @ wq^T        [4096 x 3072]
  gemm_bf16<<<dim3(CM / 128, (CH * CQK) / 64, 1), dim3(256), 0, stream>>>(
      xb, wqb, (void*)qb, CH * CQK, CDIM, CDIM, CDIM, CH * CQK, 0, 0LL, 0LL, 0LL);
  // kv = x @ wkv_a^T    [4096 x 576]
  gemm_bf16<<<dim3(CM / 128, (CKVR + CROPE) / 64, 1), dim3(256), 0, stream>>>(
      xb, wkvab, (void*)kvb, CKVR + CROPE, CDIM, CDIM, CDIM, CKVR + CROPE, 0, 0LL, 0LL, 0LL);

  prep_q<<<dim3((CM * CH) / 256), dim3(256), 0, stream>>>(qb, cosT, sinT, qnope, qpe);
  prep_kv<<<dim3(CM), dim3(256), 0, stream>>>(kvb, kvw, cosT, sinT, ckv, kpe);

  // q_lat[h] = q_nope[h] @ wbkT[h]^T   (NT, batched over heads)
  gemm_bf16<<<dim3(CM / 128, CKVR / 64, CH), dim3(256), 0, stream>>>(
      qnope, wbkT, (void*)qlat, CKVR, CNOPE, CNOPE, CNOPE, CKVR, 0,
      (long long)CM * CNOPE, (long long)CKVR * CNOPE, (long long)CM * CKVR);

  // fused causal attention over latent KV
  mla_attention<<<dim3(CS / 16, CH, CB), dim3(128), 0, stream>>>(
      qlat, qpe, ckv, kpe, olat);

  // o[h] = o_lat[h] @ wb_v[h]^T -> interleaved [m][h*128+d]
  gemm_bf16<<<dim3(CM / 128, CVDIM / 64, CH), dim3(256), 0, stream>>>(
      olat, wkvbb + (long long)CNOPE * CKVR, (void*)ob, CVDIM, CKVR, CKVR, CKVR,
      CH * CVDIM, 0,
      (long long)CM * CKVR, (long long)(CNOPE + CVDIM) * CKVR, (long long)CVDIM);

  // out = o @ wo^T  (f32 into d_out)
  gemm_bf16<<<dim3(CM / 128, CDIM / 64, 1), dim3(256), 0, stream>>>(
      ob, wob, d_out, CDIM, CH * CVDIM, CH * CVDIM, CH * CVDIM, CDIM, 1,
      0LL, 0LL, 0LL);
}
